// QNetworkLSTM_11424613007957
// MI455X (gfx1250) — matrix operand order repaired and checked by reading, not compiled, not run on hardware
//
#include <hip/hip_runtime.h>
#include <hip/hip_bf16.h>
#include <stdint.h>

// ---------------- CDNA5 WMMA / TDM types ----------------
typedef __attribute__((ext_vector_type(16))) __bf16 v16bf;
typedef __attribute__((ext_vector_type(8)))  float  v8f;
typedef __attribute__((ext_vector_type(4)))  unsigned int v4u;
typedef __attribute__((ext_vector_type(8)))  int v8i_;
typedef __attribute__((ext_vector_type(4)))  int v4i_;

#if __has_builtin(__builtin_amdgcn_tensor_load_to_lds) && __has_builtin(__builtin_amdgcn_s_wait_tensorcnt)
#define HAVE_TDM 1
#else
#define HAVE_TDM 0
#endif

// ---------------- problem constants ----------------
static constexpr int B_    = 16384;
static constexpr int NROI_ = 30;
static constexpr int T_    = 10;
static constexpr int FC_   = 32;
static constexpr int NH_   = 256;
static constexpr int NTOT_ = 427;
static constexpr int NROIF_ = NROI_ * T_;    // 300
static constexpr int NOTH_ = NTOT_ - NROIF_; // 127
static constexpr int ACT_  = 31;
static constexpr int COMB_ = NROI_ * FC_ + NH_; // 1216

// ---------------- helpers ----------------
__device__ __forceinline__ unsigned short f2bf(float f) {
  union { float f; unsigned int u; } v; v.f = f;
  unsigned int r = v.u + 0x7FFFu + ((v.u >> 16) & 1u); // RNE
  return (unsigned short)(r >> 16);
}
__device__ __forceinline__ __bf16 bits2bf(unsigned short s) {
  return __builtin_bit_cast(__bf16, s);
}
__device__ __forceinline__ float bf2f(unsigned short s) {
  union { unsigned int u; float f; } v; v.u = ((unsigned int)s) << 16;
  return v.f;
}
__device__ __forceinline__ float sigmoidf_(float x) {
  return 1.0f / (1.0f + __expf(-x));
}

#if HAVE_TDM
// Issue one TDM load of a 2-D bf16 tile (tileX contiguous elems per row,
// tileY rows, global row pitch strideX elems) into LDS at ldsOff.
// Reads past (tensX, tensY) return zero -> free K/N zero padding.
// tileY == 0 means a 1-D line.
__device__ __forceinline__ void tdm_load_tile_bf16(
    unsigned ldsOff, const unsigned short* gptr,
    unsigned tileX, unsigned tileY,
    long tensX, long tensY, long strideX)
{
  if (tensX < 0) tensX = 0;
  if (tensY < 0) tensY = 0;
  unsigned long long ga = (unsigned long long)(uintptr_t)gptr;
  unsigned long long t0 = (unsigned long long)tensX;
  unsigned long long t1 = (unsigned long long)tensY;
  unsigned long long s0 = (unsigned long long)strideX;
  v4u g0;
  g0[0] = 1u;                                   // count=1, user descriptor
  g0[1] = ldsOff;                               // lds_addr
  g0[2] = (unsigned)(ga & 0xffffffffu);         // global_addr[31:0]
  g0[3] = (unsigned)((ga >> 32) & 0x01ffffffu)  // global_addr[56:32]
        | (2u << 30);                           // type=2 ("image")
  v8i_ g1;
  g1[0] = (int)(1u << 16);                      // data_size=1 -> 2 bytes
  g1[1] = (int)((t0 & 0xffffu) << 16);          // tensor_dim0[15:0]
  g1[2] = (int)(((t0 >> 16) & 0xffffu) | ((t1 & 0xffffu) << 16));
  g1[3] = (int)(((t1 >> 16) & 0xffffu) | ((unsigned long long)tileX << 16));
  g1[4] = (int)(tileY & 0xffffu);               // tile_dim1 (tile_dim2=0)
  g1[5] = (int)(s0 & 0xffffffffu);              // tensor_dim0_stride[31:0]
  g1[6] = (int)((s0 >> 32) & 0xffffu);          // stride[47:32]; dim1_stride=0
  g1[7] = 0;
  v4i_ gz = {0, 0, 0, 0};
#if defined(__clang_major__) && (__clang_major__ >= 23)
  v8i_ gz8 = {0, 0, 0, 0, 0, 0, 0, 0};
  __builtin_amdgcn_tensor_load_to_lds(g0, g1, gz, gz, gz8, 0);
#else
  __builtin_amdgcn_tensor_load_to_lds(g0, g1, gz, gz, 0);
#endif
}
#endif // HAVE_TDM

// ============================================================
// f32 -> bf16 conversion (strided rows), grid-stride loop
// ============================================================
__global__ __launch_bounds__(256)
void cvt_bf16(const float* __restrict__ src, int srcStride,
              unsigned short* __restrict__ dst, int cols, long total)
{
  for (long i = (long)blockIdx.x * 256 + threadIdx.x; i < total;
       i += (long)gridDim.x * 256) {
    long r = i / cols;
    int  c = (int)(i - r * cols);
    dst[i] = f2bf(src[r * (long)srcStride + c]);
  }
}

// ============================================================
// C[M,N] = act(A[M,K](bf16) @ W[N,K](bf16)^T + bias)
// 64x128 tile, 8 waves; A/W tiles DMA'd into LDS by the TDM,
// double-buffered so tensor_load(k+1) overlaps WMMA(k).
// outBf16Relu: 1 -> bf16 store + ReLU, 0 -> f32 store, no act.
// ============================================================
__global__ __launch_bounds__(256)
void gemm_bf16(const unsigned short* __restrict__ A, int lda,
               const unsigned short* __restrict__ W, int ldw,
               const float* __restrict__ bias,
               void* __restrict__ Cout, int ldc,
               int M, int N, int K, int outBf16Relu)
{
  __shared__ unsigned short sA[2][64 * 32];
  __shared__ unsigned short sW[2][128 * 32];

  const int tid  = threadIdx.x;
  const int wave = tid >> 5;
  const int lane = tid & 31;
  const int half = lane >> 4;
  const int lm   = lane & 15;

  const int m0 = blockIdx.x * 64;
  const int n0 = blockIdx.y * 128;

  v8f acc[4] = {{}, {}, {}, {}};
  const int KC = (K + 31) >> 5;

#if HAVE_TDM
  if (wave == 0) { // single wave drives the TDM; TENSORcnt is per-wave
    tdm_load_tile_bf16((unsigned)(uintptr_t)&sA[0][0], A + (size_t)m0 * lda,
                       32, 64, K, M - m0, lda);
    tdm_load_tile_bf16((unsigned)(uintptr_t)&sW[0][0], W + (size_t)n0 * ldw,
                       32, 128, K, N - n0, ldw);
  }
#endif

  for (int kc = 0; kc < KC; ++kc) {
    const int cur = kc & 1;
#if HAVE_TDM
    if (wave == 0) {
      if (kc + 1 < KC) {
        const int kB = (kc + 1) << 5;
        tdm_load_tile_bf16((unsigned)(uintptr_t)&sA[cur ^ 1][0],
                           A + (size_t)m0 * lda + kB, 32, 64,
                           K - kB, M - m0, lda);
        tdm_load_tile_bf16((unsigned)(uintptr_t)&sW[cur ^ 1][0],
                           W + (size_t)n0 * ldw + kB, 32, 128,
                           K - kB, N - n0, ldw);
        __builtin_amdgcn_s_wait_tensorcnt(2); // chunk kc landed (in-order)
      } else {
        __builtin_amdgcn_s_wait_tensorcnt(0);
      }
    }
#else
    {
      const int kB = kc << 5;
      for (int i = tid; i < 64 * 32; i += 256) {
        int r = i >> 5, kk = i & 31;
        int m = m0 + r, k = kB + kk;
        sA[cur][i] = (m < M && k < K) ? A[(size_t)m * lda + k] : (unsigned short)0;
      }
      for (int i = tid; i < 128 * 32; i += 256) {
        int r = i >> 5, kk = i & 31;
        int n = n0 + r, k = kB + kk;
        sW[cur][i] = (n < N && k < K) ? W[(size_t)n * ldw + k] : (unsigned short)0;
      }
    }
#endif
    __syncthreads();

    v16bf bfrag;
    {
      int n = wave * 16 + lm;
      #pragma unroll
      for (int e = 0; e < 16; ++e) {
        int k = e + 16 * half;                 // B layout: K pairs, lane halves
        bfrag[e] = bits2bf(sW[cur][n * 32 + k]);
      }
    }
    #pragma unroll
    for (int ms = 0; ms < 4; ++ms) {
      v16bf a;
      int m = ms * 16 + lm;
      #pragma unroll
      for (int e = 0; e < 16; ++e) {
        int k = e + 16 * half;                 // contiguous K order, matches B
        a[e] = bits2bf(sA[cur][m * 32 + k]);
      }
      acc[ms] = __builtin_amdgcn_wmma_f32_16x16x32_bf16(
          false, a, false, bfrag, (short)0, acc[ms], false, false);
    }
    __syncthreads();
  }

  // epilogue: C/D layout row = r + 8*half, col = lane%16
  const int n = n0 + wave * 16 + lm;
  if (n < N) {
    const float bv = bias ? bias[n] : 0.0f;
    #pragma unroll
    for (int ms = 0; ms < 4; ++ms) {
      #pragma unroll
      for (int r = 0; r < 8; ++r) {
        int m = m0 + ms * 16 + r + 8 * half;
        if (m < M) {
          float v = acc[ms][r] + bv;
          if (outBf16Relu) {
            v = fmaxf(v, 0.0f);
            ((unsigned short*)Cout)[(size_t)m * ldc + n] = f2bf(v);
          } else {
            ((float*)Cout)[(size_t)m * ldc + n] = v;
          }
        }
      }
    }
  }
}

// ============================================================
// Bidirectional masked LSTM (H=64, T=10) + FC(128->32) + ReLU.
// One wave owns 16 sequences; W_hh (bf16) TDM'd into shared LDS.
// ============================================================
__global__ __launch_bounds__(128)
void lstm_bidir_fc(const float* __restrict__ x,
                   const unsigned short* __restrict__ Whhf_bf,
                   const unsigned short* __restrict__ Whhb_bf,
                   const float* __restrict__ Wih_f, const float* __restrict__ bih_f,
                   const float* __restrict__ bhh_f,
                   const float* __restrict__ Wih_b, const float* __restrict__ bih_b,
                   const float* __restrict__ bhh_b,
                   const unsigned short* __restrict__ Wfc_bf,
                   const float* __restrict__ bfc,
                   unsigned short* __restrict__ combined) // [B, 1216] bf16
{
  __shared__ unsigned short sWhh[256 * 64];
  __shared__ float sWih[256];
  __shared__ float sBias[256];
  __shared__ unsigned short sWfc[32 * 128];
  __shared__ float sG[4][16][256];
  __shared__ unsigned short sH[4][16][64];
  __shared__ float sC[4][16][64];
  __shared__ float sX[4][16][T_];
  __shared__ int   sLen[4][16];
  __shared__ unsigned short sHcomb[4][16][128];

  const int tid  = threadIdx.x;
  const int wave = tid >> 5;
  const int lane = tid & 31;
  const int half = lane >> 4;
  const int lm   = lane & 15;

  const int seqBase = blockIdx.x * 64 + wave * 16;

#if HAVE_TDM
  if (wave == 0)   // 1-D line: 32x128 bf16 FC weights
    tdm_load_tile_bf16((unsigned)(uintptr_t)&sWfc[0], Wfc_bf,
                       32 * 128, 0, 32 * 128, 1, 32 * 128);
#else
  for (int i = tid; i < 32 * 128; i += 128) sWfc[i] = Wfc_bf[i];
#endif

  // this wave's 16 sequences and their lengths
  for (int i = lane; i < 16 * T_; i += 32) {
    int m = i / T_, t = i % T_;
    int s = seqBase + m;
    int b = s / NROI_, r = s % NROI_;
    sX[wave][m][t] = x[(size_t)b * NTOT_ + r * T_ + t];
  }
  __builtin_amdgcn_wave_barrier();
  if (lane < 16) {
    int cnt = 0;
    #pragma unroll
    for (int t = 0; t < T_; ++t) cnt += (sX[wave][lane][t] != 0.0f) ? 1 : 0;
    sLen[wave][lane] = cnt;
  }

  for (int dir = 0; dir < 2; ++dir) {
    const unsigned short* Whh = dir ? Whhb_bf : Whhf_bf;
    const float* Wih = dir ? Wih_b : Wih_f;
    const float* bi  = dir ? bih_b : bih_f;
    const float* bh  = dir ? bhh_b : bhh_f;

    __syncthreads(); // everyone done with previous direction's weights
#if HAVE_TDM
    if (wave == 0) {
      tdm_load_tile_bf16((unsigned)(uintptr_t)&sWhh[0], Whh,
                         256 * 64, 0, 256 * 64, 1, 256 * 64);
      __builtin_amdgcn_s_wait_tensorcnt(0); // also covers the Wfc load
    }
#else
    for (int i = tid; i < 256 * 64; i += 128) sWhh[i] = Whh[i];
#endif
    for (int i = tid; i < 256; i += 128) { sWih[i] = Wih[i]; sBias[i] = bi[i] + bh[i]; }
    __syncthreads();

    for (int i = lane; i < 16 * 64; i += 32) {
      sH[wave][i >> 6][i & 63] = (unsigned short)0;
      sC[wave][i >> 6][i & 63] = 0.0f;
    }
    __builtin_amdgcn_wave_barrier();

    for (int t = 0; t < T_; ++t) {
      const int rt = dir ? (T_ - 1 - t) : t;

      v16bf a0, a1;
      #pragma unroll
      for (int e = 0; e < 16; ++e) {
        int k = e + 16 * half;
        a0[e] = bits2bf(sH[wave][lm][k]);
        a1[e] = bits2bf(sH[wave][lm][k + 32]);
      }

      for (int nt = 0; nt < 16; ++nt) {
        v16bf b0, b1;
        const int n = nt * 16 + lm;
        #pragma unroll
        for (int e = 0; e < 16; ++e) {
          int k = e + 16 * half;
          b0[e] = bits2bf(sWhh[n * 64 + k]);
          b1[e] = bits2bf(sWhh[n * 64 + k + 32]);
        }
        v8f acc = {};
        acc = __builtin_amdgcn_wmma_f32_16x16x32_bf16(false, a0, false, b0,
                                                      (short)0, acc, false, false);
        acc = __builtin_amdgcn_wmma_f32_16x16x32_bf16(false, a1, false, b1,
                                                      (short)0, acc, false, false);
        #pragma unroll
        for (int r = 0; r < 8; ++r)
          sG[wave][r + 8 * half][nt * 16 + lm] = acc[r];
      }
      __builtin_amdgcn_wave_barrier();

      for (int i = lane; i < 16 * 64; i += 32) {
        const int m = i >> 6, j = i & 63;
        const float xv = sX[wave][m][rt];
        const float gi = sG[wave][m][j]       + xv * sWih[j]       + sBias[j];
        const float gf = sG[wave][m][j + 64]  + xv * sWih[j + 64]  + sBias[j + 64];
        const float gg = sG[wave][m][j + 128] + xv * sWih[j + 128] + sBias[j + 128];
        const float go = sG[wave][m][j + 192] + xv * sWih[j + 192] + sBias[j + 192];
        const float c  = sC[wave][m][j];
        const float cn = sigmoidf_(gf) * c + sigmoidf_(gi) * tanhf(gg);
        const float hn = sigmoidf_(go) * tanhf(cn);
        const bool valid = rt < sLen[wave][m];
        const float hprev = bf2f(sH[wave][m][j]);
        sC[wave][m][j] = valid ? cn : c;
        sH[wave][m][j] = f2bf(valid ? hn : hprev);
      }
      __builtin_amdgcn_wave_barrier();
    }

    for (int i = lane; i < 16 * 64; i += 32)
      sHcomb[wave][i >> 6][dir * 64 + (i & 63)] = sH[wave][i >> 6][i & 63];
    __builtin_amdgcn_wave_barrier();
  }

  // FC: relu(h_comb[16,128] @ Wfc^T[128,32] + b_fc)
  v8f accs[2] = {{}, {}};
  #pragma unroll
  for (int kc = 0; kc < 4; ++kc) {
    v16bf a;
    #pragma unroll
    for (int e = 0; e < 16; ++e) {
      int k = e + 16 * half + 32 * kc;
      a[e] = bits2bf(sHcomb[wave][lm][k]);
    }
    #pragma unroll
    for (int nt = 0; nt < 2; ++nt) {
      v16bf bfr;
      const int n = nt * 16 + lm;
      #pragma unroll
      for (int e = 0; e < 16; ++e) {
        int k = e + 16 * half + 32 * kc;
        bfr[e] = bits2bf(sWfc[n * 128 + k]);
      }
      accs[nt] = __builtin_amdgcn_wmma_f32_16x16x32_bf16(
          false, a, false, bfr, (short)0, accs[nt], false, false);
    }
  }
  #pragma unroll
  for (int nt = 0; nt < 2; ++nt) {
    const int n = nt * 16 + lm;
    #pragma unroll
    for (int r = 0; r < 8; ++r) {
      const int m = r + 8 * half;
      const int s = seqBase + m;
      const int b = s / NROI_, rr = s % NROI_;
      float v = accs[nt][r] + bfc[n];
      combined[(size_t)b * COMB_ + rr * FC_ + n] = f2bf(fmaxf(v, 0.0f));
    }
  }
}

// ============================================================
// launcher
// ============================================================
extern "C" void kernel_launch(void* const* d_in, const int* in_sizes, int n_in,
                              void* d_out, int out_size, void* d_ws, size_t ws_size,
                              hipStream_t stream) {
  (void)in_sizes; (void)n_in; (void)out_size; (void)ws_size;

  const float* x     = (const float*)d_in[0];
  const float* Wih_f = (const float*)d_in[1];
  const float* Whh_f = (const float*)d_in[2];
  const float* bih_f = (const float*)d_in[3];
  const float* bhh_f = (const float*)d_in[4];
  const float* Wih_b = (const float*)d_in[5];
  const float* Whh_b = (const float*)d_in[6];
  const float* bih_b = (const float*)d_in[7];
  const float* bhh_b = (const float*)d_in[8];
  const float* Wfc   = (const float*)d_in[9];
  const float* bfc   = (const float*)d_in[10];
  const float* W1    = (const float*)d_in[11];
  const float* b1    = (const float*)d_in[12];
  const float* W2    = (const float*)d_in[13];
  const float* b2    = (const float*)d_in[14];
  const float* W3    = (const float*)d_in[15];
  const float* b3    = (const float*)d_in[16];
  const float* W_out = (const float*)d_in[17];
  const float* b_out = (const float*)d_in[18];
  float* out = (float*)d_out;

  // bf16 scratch layout
  unsigned short* p = (unsigned short*)d_ws;
  unsigned short* xob   = p; p += (size_t)B_ * NOTH_;   // [B,127]
  unsigned short* h1b   = p; p += (size_t)B_ * NH_;
  unsigned short* h2b   = p; p += (size_t)B_ * NH_;
  unsigned short* combb = p; p += (size_t)B_ * COMB_;
  unsigned short* W1b   = p; p += (size_t)NH_ * NOTH_;
  unsigned short* W2b   = p; p += (size_t)NH_ * NH_;
  unsigned short* W3b   = p; p += (size_t)NH_ * NH_;
  unsigned short* Woutb = p; p += (size_t)ACT_ * COMB_;
  unsigned short* Whhfb = p; p += 256 * 64;
  unsigned short* Whhbb = p; p += 256 * 64;
  unsigned short* Wfcb  = p; p += 32 * 128;

  const dim3 blk(256);
  auto cvtGrid = [](long total) {
    long g = (total + 255) / 256;
    return dim3((unsigned)(g > 4096 ? 4096 : g));
  };

  // one-pass f32 -> bf16 conversions (activations input + all weights)
  cvt_bf16<<<cvtGrid((long)B_ * NOTH_), blk, 0, stream>>>(x + NROIF_, NTOT_, xob, NOTH_, (long)B_ * NOTH_);
  cvt_bf16<<<cvtGrid((long)NH_ * NOTH_), blk, 0, stream>>>(W1, NOTH_, W1b, NOTH_, (long)NH_ * NOTH_);
  cvt_bf16<<<cvtGrid((long)NH_ * NH_), blk, 0, stream>>>(W2, NH_, W2b, NH_, (long)NH_ * NH_);
  cvt_bf16<<<cvtGrid((long)NH_ * NH_), blk, 0, stream>>>(W3, NH_, W3b, NH_, (long)NH_ * NH_);
  cvt_bf16<<<cvtGrid((long)ACT_ * COMB_), blk, 0, stream>>>(W_out, COMB_, Woutb, COMB_, (long)ACT_ * COMB_);
  cvt_bf16<<<cvtGrid(256 * 64), blk, 0, stream>>>(Whh_f, 64, Whhfb, 64, 256 * 64);
  cvt_bf16<<<cvtGrid(256 * 64), blk, 0, stream>>>(Whh_b, 64, Whhbb, 64, 256 * 64);
  cvt_bf16<<<cvtGrid(32 * 128), blk, 0, stream>>>(Wfc, 128, Wfcb, 128, 32 * 128);

  // other-branch MLP (TDM-staged, double-buffered WMMA)
  gemm_bf16<<<dim3(B_ / 64, 2), blk, 0, stream>>>(
      xob, NOTH_, W1b, NOTH_, b1, h1b, NH_, B_, NH_, NOTH_, 1);
  gemm_bf16<<<dim3(B_ / 64, 2), blk, 0, stream>>>(
      h1b, NH_, W2b, NH_, b2, h2b, NH_, B_, NH_, NH_, 1);
  gemm_bf16<<<dim3(B_ / 64, 2), blk, 0, stream>>>(
      h2b, NH_, W3b, NH_, b3, combb + NROI_ * FC_, COMB_, B_, NH_, NH_, 1);

  // bidirectional LSTM + FC -> combined[:, 0:960]
  lstm_bidir_fc<<<dim3((B_ * NROI_) / 64), dim3(128), 0, stream>>>(
      x, Whhfb, Whhbb, Wih_f, bih_f, bhh_f, Wih_b, bih_b, bhh_b, Wfcb, bfc, combb);

  // output head: q = combined @ W_out^T + b_out (f32 out)
  gemm_bf16<<<dim3(B_ / 64, 1), blk, 0, stream>>>(
      combb, COMB_, Woutb, COMB_, b_out, out, ACT_, B_, ACT_, COMB_, 0);
}
